// AdaptiveJacobianPrunedViT_24361054503097
// MI455X (gfx1250) — matrix operand order repaired and checked by reading, not compiled, not run on hardware
//
#include <hip/hip_runtime.h>
#include <hip/hip_bf16.h>
#include <math.h>

// ---------------- model constants ----------------
#define LNUM    12
#define DMODEL  384
#define NHEAD   6
#define HDIM    64
#define NPATCH  196
#define NTOK    197          // NPATCH + cls
#define TCAP    208          // padded token capacity (13 * 16)
#define KCAP    224          // padded K-dim for P@V (7 * 32)
#define BATCH   64
#define NCLS    1000
#define QK_SCALE 0.125f      // 64^-0.5
#define LEPS    1e-6f

typedef __attribute__((ext_vector_type(16))) _Float16 v16h;
typedef __attribute__((ext_vector_type(8)))  float    v8f;

union F8  { v8f  v; float    f[8];  };
union H16 { v16h v; _Float16 h[16]; unsigned u[8]; };

struct Scalars { int n_tok; int n_tok_new; float prev_mass; };

// =====================================================================
// fp32 -> fp16 bulk convert (weights + pixels, run once per launch)
// =====================================================================
__global__ void k_cvt(const float* __restrict__ s, _Float16* __restrict__ d, int n)
{
    int i = blockIdx.x * blockDim.x + threadIdx.x;
    if (i < n) d[i] = (_Float16)s[i];
}
static void launch_cvt(const float* s, _Float16* d, int n, hipStream_t st)
{
    k_cvt<<<(n + 255) / 256, 256, 0, st>>>(s, d, n);
}

// =====================================================================
// WMMA GEMM: C[M,N] = act(A[M,K] @ W[N,K]^T + bias) (+ R)
// A, W fp16; fp32 accumulate. One wave computes a 16x64 strip
// (4 accumulators); A fragment feeds 4 WMMAs per k-step.
// ACT: 0 none, 1 exact GELU.  O16: fp16 C output.  EXACT: M%16==0 && N%64==0.
// =====================================================================
template<int ACT, int O16, int EXACT>
__global__ void k_gemm(const _Float16* __restrict__ A, const _Float16* __restrict__ W,
                       const float* __restrict__ bias, const float* __restrict__ R,
                       void* __restrict__ Cv, int M, int N, int K)
{
    int wid  = (blockIdx.x * blockDim.x + threadIdx.x) >> 5;
    int lane = threadIdx.x & 31;
    int mtiles  = (M + 15) >> 4;
    int ngroups = (N + 63) >> 6;
    if (wid >= mtiles * ngroups) return;
    int mt = wid / ngroups, ng = wid % ngroups;
    int m  = lane & 15, kh = lane >> 4;

    int arow = mt * 16 + m; if (!EXACT && arow > M - 1) arow = M - 1;
    const _Float16* Ar = A + (size_t)arow * K;
    const _Float16* Wr[4];
#pragma unroll
    for (int t = 0; t < 4; ++t) {
        int bc = ng * 64 + t * 16 + m; if (!EXACT && bc > N - 1) bc = N - 1;
        Wr[t] = W + (size_t)bc * K;
    }

    F8 acc[4];
#pragma unroll
    for (int t = 0; t < 4; ++t) acc[t].v = (v8f){0.f,0.f,0.f,0.f,0.f,0.f,0.f,0.f};

    for (int kk = 0; kk < K; kk += 32) {
        if (kk + 32 < K) __builtin_prefetch(Ar + kk + 32, 0, 0);
        H16 a;
#pragma unroll
        for (int v = 0; v < 8; ++v) {
            int ka = kk + ((v >> 2) << 4) + (kh << 3) + ((v & 3) << 1);
            a.u[v] = *(const unsigned*)(Ar + ka);
        }
#pragma unroll
        for (int t = 0; t < 4; ++t) {
            H16 b;
#pragma unroll
            for (int v = 0; v < 8; ++v) {
                int kb = kk + ((v >> 2) << 4) + (kh << 3) + ((v & 3) << 1);
                b.u[v] = *(const unsigned*)(Wr[t] + kb);
            }
            acc[t].v = __builtin_amdgcn_wmma_f32_16x16x32_f16(
                false, a.v, false, b.v, (short)0, acc[t].v, false, false);
        }
    }

#pragma unroll
    for (int t = 0; t < 4; ++t) {
        int n = ng * 64 + t * 16 + m;
        if (EXACT || n < N) {
            float bn = bias ? bias[n] : 0.f;
#pragma unroll
            for (int r = 0; r < 8; ++r) {
                int mm = mt * 16 + r + (kh << 3);
                if (EXACT || mm < M) {
                    float xv = acc[t].f[r] + bn;
                    if (R) xv += R[(size_t)mm * N + n];
                    if (ACT == 1) xv = 0.5f * xv * (1.f + erff(xv * 0.70710678118654752f));
                    if (O16) ((_Float16*)Cv)[(size_t)mm * N + n] = (_Float16)xv;
                    else     ((float*)Cv)[(size_t)mm * N + n] = xv;
                }
            }
        }
    }
}

template<int ACT, int O16, int EXACT>
static void launch_gemm(const _Float16* A, const _Float16* W, const float* bias,
                        const float* R, void* C, int M, int N, int K,
                        hipStream_t stream)
{
    int waves  = ((M + 15) / 16) * ((N + 63) / 64);
    int blocks = (waves + 7) / 8;
    k_gemm<ACT, O16, EXACT><<<blocks, 256, 0, stream>>>(A, W, bias, R, C, M, N, K);
}

// =====================================================================
// Patch embedding: im2col-gathered GEMM per image, fp16 pixels.
// M = 196 (13 tiles), N = 384 (6 groups of 64), K = 768. Output fp32 xt.
// =====================================================================
__global__ void k_patch_embed(const _Float16* __restrict__ xh, const _Float16* __restrict__ pw,
                              const float* __restrict__ pb, const float* __restrict__ pos,
                              float* __restrict__ xt)
{
    const int MT = 13, NG = 6;
    int wid = (blockIdx.x * blockDim.x + threadIdx.x) >> 5;
    if (wid >= BATCH * MT * NG) return;
    int b  = wid / (MT * NG);
    int mt = (wid / NG) % MT;
    int ng = wid % NG;
    int lane = threadIdx.x & 31;
    int m = lane & 15, kh = lane >> 4;

    int p = mt * 16 + m; if (p > NPATCH - 1) p = NPATCH - 1;
    int py = p / 14, px = p % 14;
    const _Float16* xb = xh + ((size_t)b * 3) * 224 * 224 + (size_t)(py * 16) * 224 + px * 16;
    const _Float16* Wr[4];
#pragma unroll
    for (int t = 0; t < 4; ++t) Wr[t] = pw + (size_t)(ng * 64 + t * 16 + m) * 768;

    F8 acc[4];
#pragma unroll
    for (int t = 0; t < 4; ++t) acc[t].v = (v8f){0.f,0.f,0.f,0.f,0.f,0.f,0.f,0.f};

    for (int kk = 0; kk < 768; kk += 32) {
        H16 a;
#pragma unroll
        for (int v = 0; v < 8; ++v) {
            int k = kk + ((v >> 2) << 4) + (kh << 3) + ((v & 3) << 1);
            int c = k >> 8, rem = k & 255, iy = rem >> 4, ix = rem & 15;
            // pair (k, k+1) is contiguous along ix
            a.u[v] = *(const unsigned*)(xb + ((size_t)c * 224 + iy) * 224 + ix);
        }
#pragma unroll
        for (int t = 0; t < 4; ++t) {
            H16 bf;
#pragma unroll
            for (int v = 0; v < 8; ++v) {
                int kb = kk + ((v >> 2) << 4) + (kh << 3) + ((v & 3) << 1);
                bf.u[v] = *(const unsigned*)(Wr[t] + kb);
            }
            acc[t].v = __builtin_amdgcn_wmma_f32_16x16x32_f16(
                false, a.v, false, bf.v, (short)0, acc[t].v, false, false);
        }
    }

#pragma unroll
    for (int t = 0; t < 4; ++t) {
        int n = ng * 64 + t * 16 + m;
#pragma unroll
        for (int r = 0; r < 8; ++r) {
            int pp = mt * 16 + r + (kh << 3);
            if (pp < NPATCH) {
                float v = acc[t].f[r] + pb[n] + pos[(size_t)(1 + pp) * DMODEL + n];
                xt[((size_t)b * TCAP + 1 + pp) * DMODEL + n] = v;
            }
        }
    }
}

__global__ void k_cls_init(const float* __restrict__ cls, const float* __restrict__ pos,
                           float* __restrict__ xt)
{
    int b = blockIdx.x;
    for (int d = threadIdx.x; d < DMODEL; d += blockDim.x)
        xt[(size_t)b * TCAP * DMODEL + d] = cls[d] + pos[d];
}

// =====================================================================
// LayerNorm over last dim (384), fp32 in -> fp16 out.
// =====================================================================
__global__ void k_ln(const float* __restrict__ X, const float* __restrict__ w,
                     const float* __restrict__ bs, _Float16* __restrict__ Y,
                     int xstride, int ystride)
{
    __shared__ float red[128];
    int t = blockIdx.x;
    const float* xr = X + (size_t)t * xstride;
    _Float16* yr = Y + (size_t)t * ystride;

    float s = 0.f;
    for (int d = threadIdx.x; d < DMODEL; d += 128) s += xr[d];
    red[threadIdx.x] = s; __syncthreads();
    for (int o = 64; o > 0; o >>= 1) {
        if (threadIdx.x < o) red[threadIdx.x] += red[threadIdx.x + o];
        __syncthreads();
    }
    float mu = red[0] / DMODEL; __syncthreads();

    float v = 0.f;
    for (int d = threadIdx.x; d < DMODEL; d += 128) { float z = xr[d] - mu; v += z * z; }
    red[threadIdx.x] = v; __syncthreads();
    for (int o = 64; o > 0; o >>= 1) {
        if (threadIdx.x < o) red[threadIdx.x] += red[threadIdx.x + o];
        __syncthreads();
    }
    float inv = rsqrtf(red[0] / DMODEL + LEPS);

    for (int d = threadIdx.x; d < DMODEL; d += 128)
        yr[d] = (_Float16)((xr[d] - mu) * inv * w[d] + bs[d]);
}

// =====================================================================
// Repack fused QKV fp16 [B*TCAP,1152] into:
//   qf, kf : [B,H,KCAP,64]   (row-major tokens)
//   vt     : [B,H,64,KCAP]   (transposed: contiguous along tokens)
// =====================================================================
__global__ void k_repack(const _Float16* __restrict__ qkv, _Float16* __restrict__ qf,
                         _Float16* __restrict__ kf, _Float16* __restrict__ vt)
{
    size_t idx = (size_t)blockIdx.x * blockDim.x + threadIdx.x;
    const size_t total = (size_t)BATCH * TCAP * 1152;
    if (idx >= total) return;
    int c = (int)(idx % 1152);
    size_t r = idx / 1152;
    int t = (int)(r % TCAP);
    int b = (int)(r / TCAP);
    int which = c / DMODEL;
    int hc = c % DMODEL;
    int h = hc >> 6, d = hc & 63;
    _Float16 val = qkv[idx];
    size_t bh = (size_t)b * NHEAD + h;
    if (which == 0)      qf[(bh * KCAP + t) * HDIM + d] = val;
    else if (which == 1) kf[(bh * KCAP + t) * HDIM + d] = val;
    else                 vt[(bh * HDIM + d) * KCAP + t] = val;
}

// =====================================================================
// Attention: one wave per (b,h,q-tile). WMMA for Q@K^T and P@V,
// fp32 softmax through LDS, key masking against dynamic token count.
// V is pre-transposed so both WMMA operand streams are dword loads.
// Output fp16 O [B,TCAP,DMODEL].
// =====================================================================
__global__ void __launch_bounds__(32)
k_attn(const _Float16* __restrict__ qf, const _Float16* __restrict__ kf,
       const _Float16* __restrict__ vt, _Float16* __restrict__ O,
       const int* __restrict__ ntokp)
{
    __shared__ float    scb[16 * TCAP];
    __shared__ _Float16 pbb[16 * KCAP];

    int bh = blockIdx.x;          // b*6 + h
    int qt = blockIdx.y;          // 0..12
    int lane = threadIdx.x;
    int m = lane & 15, kh = lane >> 4;
    int ntok = *ntokp;

    const _Float16* Q  = qf + (size_t)bh * KCAP * HDIM;
    const _Float16* Km = kf + (size_t)bh * KCAP * HDIM;
    const _Float16* Vt = vt + (size_t)bh * HDIM * KCAP;
    int qrow = qt * 16 + m;

    // ---- phase 1: S = Q K^T * scale (13 key tiles, K=64 -> 2 wmma each)
    for (int jt = 0; jt < 13; ++jt) {
        F8 acc; acc.v = (v8f){0.f,0.f,0.f,0.f,0.f,0.f,0.f,0.f};
        int ktok = jt * 16 + m;
#pragma unroll
        for (int kk = 0; kk < HDIM; kk += 32) {
            H16 a, b;
#pragma unroll
            for (int v = 0; v < 8; ++v) {
                int ka = kk + ((v >> 2) << 4) + (kh << 3) + ((v & 3) << 1);
                a.u[v] = *(const unsigned*)(Q + (size_t)qrow * HDIM + ka);
                int kb = kk + ((v >> 2) << 4) + (kh << 3) + ((v & 3) << 1);
                b.u[v] = *(const unsigned*)(Km + (size_t)ktok * HDIM + kb);
            }
            acc.v = __builtin_amdgcn_wmma_f32_16x16x32_f16(
                false, a.v, false, b.v, (short)0, acc.v, false, false);
        }
        int col = jt * 16 + m;
#pragma unroll
        for (int r = 0; r < 8; ++r) {
            int row = r + (kh << 3);
            scb[row * TCAP + col] = (col < ntok) ? acc.f[r] * QK_SCALE : -1e30f;
        }
    }
    __syncthreads();

    // ---- phase 2: row softmax -> fp16 P (zero-padded to KCAP)
    if (lane < 16) {
        int row = lane;
        float mx = -1e30f;
        for (int j = 0; j < TCAP; ++j) mx = fmaxf(mx, scb[row * TCAP + j]);
        float s = 0.f;
        for (int j = 0; j < TCAP; ++j) {
            float e = expf(scb[row * TCAP + j] - mx);
            scb[row * TCAP + j] = e; s += e;
        }
        float inv = 1.f / s;
        for (int j = 0; j < TCAP; ++j)
            pbb[row * KCAP + j] = (_Float16)(scb[row * TCAP + j] * inv);
        for (int j = TCAP; j < KCAP; ++j)
            pbb[row * KCAP + j] = (_Float16)0.f;
    }
    __syncthreads();

    // ---- phase 3: O = P V (K = 224 -> 7 wmma per 16-wide d tile)
    int b = bh / NHEAD, h = bh % NHEAD;
#pragma unroll
    for (int dt = 0; dt < 4; ++dt) {
        F8 acc; acc.v = (v8f){0.f,0.f,0.f,0.f,0.f,0.f,0.f,0.f};
        int dcol = dt * 16 + m;
        for (int kk = 0; kk < KCAP; kk += 32) {
            H16 a, bb;
#pragma unroll
            for (int v = 0; v < 8; ++v) {
                int ka = kk + ((v >> 2) << 4) + (kh << 3) + ((v & 3) << 1);
                a.u[v] = *(const unsigned*)&pbb[m * KCAP + ka];
                int kb = kk + ((v >> 2) << 4) + (kh << 3) + ((v & 3) << 1);
                bb.u[v] = *(const unsigned*)(Vt + (size_t)dcol * KCAP + kb);
            }
            acc.v = __builtin_amdgcn_wmma_f32_16x16x32_f16(
                false, a.v, false, bb.v, (short)0, acc.v, false, false);
        }
#pragma unroll
        for (int r = 0; r < 8; ++r) {
            int tok = qt * 16 + r + (kh << 3);
            O[((size_t)b * TCAP + tok) * DMODEL + h * HDIM + dcol] = (_Float16)acc.f[r];
        }
    }
}

// =====================================================================
// Importance: imp[b][j] = mean_h( softmax(q_cls . k_j * scale) * |v_j| )
// =====================================================================
__global__ void k_importance(const _Float16* __restrict__ qf, const _Float16* __restrict__ kf,
                             const _Float16* __restrict__ vt, float* __restrict__ imp,
                             const int* __restrict__ ntokp)
{
    __shared__ float red[256];
    int b = blockIdx.x;
    int j = threadIdx.x;
    int ntok = *ntokp;
    float accum = 0.f;

    for (int h = 0; h < NHEAD; ++h) {
        size_t bh = (size_t)b * NHEAD + h;
        const _Float16* Q  = qf + bh * KCAP * HDIM;
        const _Float16* Km = kf + bh * KCAP * HDIM;
        const _Float16* Vt = vt + bh * HDIM * KCAP;
        float sj = -1e30f, vn = 0.f;
        if (j < ntok) {
            float dot = 0.f, vs = 0.f;
            for (int d = 0; d < HDIM; ++d) {
                dot += (float)Q[d] * (float)Km[(size_t)j * HDIM + d];
                float vv = (float)Vt[(size_t)d * KCAP + j]; vs += vv * vv;
            }
            sj = dot * QK_SCALE; vn = sqrtf(vs);
        }
        red[threadIdx.x] = sj; __syncthreads();
        for (int o = 128; o > 0; o >>= 1) {
            if (threadIdx.x < o) red[threadIdx.x] = fmaxf(red[threadIdx.x], red[threadIdx.x + o]);
            __syncthreads();
        }
        float mx = red[0]; __syncthreads();
        float e = (j < ntok) ? expf(sj - mx) : 0.f;
        red[threadIdx.x] = e; __syncthreads();
        for (int o = 128; o > 0; o >>= 1) {
            if (threadIdx.x < o) red[threadIdx.x] += red[threadIdx.x + o];
            __syncthreads();
        }
        float s = red[0]; __syncthreads();
        accum += (e / s) * vn;
    }
    if (j < KCAP) imp[(size_t)b * KCAP + j] = (j < ntok) ? accum / NHEAD : 0.f;
}

// =====================================================================
// Prune decision + top-k (rank select, stable ties like jax.lax.top_k)
// =====================================================================
__global__ void k_prune(const float* __restrict__ imp, int* __restrict__ keep,
                        Scalars* __restrict__ sc)
{
    __shared__ float s[KCAP];
    __shared__ float frac[BATCH];
    int tid = threadIdx.x;
    int ntok = sc->n_tok;
    int Np = ntok - 1;

    if (tid < BATCH) {
        float tot = 0.f, part = 0.f;
        for (int j = 0; j < ntok; ++j) {
            float v = imp[(size_t)tid * KCAP + j];
            tot += v; if (j >= 1) part += v;
        }
        frac[tid] = part / (tot + LEPS);
    }
    if (tid < KCAP) {
        float m = 0.f;
        for (int b = 0; b < BATCH; ++b) m += imp[(size_t)b * KCAP + tid];
        s[tid] = m / BATCH;
    }
    __syncthreads();

    if (tid == 0) {
        int nnew = ntok;
        if (Np > 16) {                       // MIN_TOKENS
            float mass = 0.f;
            for (int b = 0; b < BATCH; ++b) mass += frac[b];
            mass /= BATCH;
            float kr = 0.5f * mass / (sc->prev_mass + LEPS);   // GAMMA=0.5
            kr = fminf(fmaxf(kr, 0.f), 1.f);
            int nsel = (int)((float)Np * kr);
            if (nsel < 16) nsel = 16;
            if (nsel < Np) {
                int cnt = 0;
                keep[cnt++] = 0;
                for (int j = 1; j < ntok; ++j) {
                    int rank = 0;
                    for (int i = 1; i < ntok; ++i)
                        if (s[i] > s[j] || (s[i] == s[j] && i < j)) rank++;
                    if (rank < nsel) keep[cnt++] = j;
                }
                nnew = cnt;
            } else {
                for (int j = 0; j < ntok; ++j) keep[j] = j;
            }
            sc->prev_mass = mass;
        } else {
            for (int j = 0; j < ntok; ++j) keep[j] = j;
        }
        sc->n_tok_new = nnew;
    }
}

__global__ void k_gather(const float* __restrict__ src, float* __restrict__ dst,
                         const int* __restrict__ keep, const Scalars* __restrict__ sc)
{
    int b = blockIdx.x, j = blockIdx.y;
    int nnew = sc->n_tok_new;
    int sidx = (j < nnew) ? keep[j] : j;
    for (int d = threadIdx.x; d < DMODEL; d += blockDim.x)
        dst[((size_t)b * TCAP + j) * DMODEL + d] =
            src[((size_t)b * TCAP + sidx) * DMODEL + d];
}

__global__ void k_commit(Scalars* sc) { sc->n_tok = sc->n_tok_new; }
__global__ void k_init(Scalars* sc)   { sc->n_tok = NTOK; sc->n_tok_new = NTOK; sc->prev_mass = 1.0f; }

// =====================================================================
// Host driver
// =====================================================================
extern "C" void kernel_launch(void* const* d_in, const int* in_sizes, int n_in,
                              void* d_out, int out_size, void* d_ws, size_t ws_size,
                              hipStream_t stream)
{
    const float* x       = (const float*)d_in[0];
    const float* patch_w = (const float*)d_in[1];
    const float* patch_b = (const float*)d_in[2];
    const float* cls_tok = (const float*)d_in[3];
    const float* pos     = (const float*)d_in[4];
    const float* ln1_w   = (const float*)d_in[5];
    const float* ln1_b   = (const float*)d_in[6];
    const float* qkv_w   = (const float*)d_in[7];
    const float* qkv_b   = (const float*)d_in[8];
    const float* proj_w  = (const float*)d_in[9];
    const float* proj_b  = (const float*)d_in[10];
    const float* ln2_w   = (const float*)d_in[11];
    const float* ln2_b   = (const float*)d_in[12];
    const float* fc1_w   = (const float*)d_in[13];
    const float* fc1_b   = (const float*)d_in[14];
    const float* fc2_w   = (const float*)d_in[15];
    const float* fc2_b   = (const float*)d_in[16];
    const float* norm_w  = (const float*)d_in[17];
    const float* norm_b  = (const float*)d_in[18];
    const float* head_w  = (const float*)d_in[19];
    const float* head_b  = (const float*)d_in[20];
    float* out = (float*)d_out;

    char* ws = (char*)d_ws;
    size_t off = 0;
    auto alloc = [&](size_t bytes) -> void* {
        off = (off + 255) & ~(size_t)255;
        void* p = ws + off;
        off += bytes;
        return p;
    };

    // fp32 residual stream (ping-pong)
    float*    xt0  = (float*)   alloc((size_t)BATCH * TCAP * DMODEL * 4);
    float*    xt1  = (float*)   alloc((size_t)BATCH * TCAP * DMODEL * 4);
    // fp16 activations
    _Float16* xnh  = (_Float16*)alloc((size_t)BATCH * TCAP * DMODEL * 2);
    _Float16* bigh = (_Float16*)alloc((size_t)BATCH * TCAP * 1536 * 2);
    _Float16* oh   = (_Float16*)alloc((size_t)BATCH * TCAP * DMODEL * 2);
    _Float16* qfp  = (_Float16*)alloc((size_t)BATCH * NHEAD * KCAP * HDIM * 2);
    _Float16* kfp  = (_Float16*)alloc((size_t)BATCH * NHEAD * KCAP * HDIM * 2);
    _Float16* vtp  = (_Float16*)alloc((size_t)BATCH * NHEAD * HDIM * KCAP * 2);
    _Float16* clsn = (_Float16*)alloc((size_t)BATCH * DMODEL * 2);
    _Float16* xhp  = (_Float16*)alloc((size_t)BATCH * 3 * 224 * 224 * 2);
    // fp16 weights
    _Float16* qkvw_h  = (_Float16*)alloc((size_t)LNUM * 1152 * DMODEL * 2);
    _Float16* projw_h = (_Float16*)alloc((size_t)LNUM * DMODEL * DMODEL * 2);
    _Float16* fc1w_h  = (_Float16*)alloc((size_t)LNUM * 1536 * DMODEL * 2);
    _Float16* fc2w_h  = (_Float16*)alloc((size_t)LNUM * DMODEL * 1536 * 2);
    _Float16* headw_h = (_Float16*)alloc((size_t)NCLS * DMODEL * 2);
    _Float16* patchw_h= (_Float16*)alloc((size_t)DMODEL * 768 * 2);
    // pruning state
    float*    imp  = (float*)   alloc((size_t)BATCH * KCAP * 4);
    int*      keep = (int*)     alloc((size_t)KCAP * 4);
    Scalars*  sc   = (Scalars*) alloc(256);

    const int Mrows = BATCH * TCAP;

    // one-time conversions (deterministic; rerun every call)
    launch_cvt(qkv_w,  qkvw_h,  LNUM * 1152 * DMODEL, stream);
    launch_cvt(proj_w, projw_h, LNUM * DMODEL * DMODEL, stream);
    launch_cvt(fc1_w,  fc1w_h,  LNUM * 1536 * DMODEL, stream);
    launch_cvt(fc2_w,  fc2w_h,  LNUM * DMODEL * 1536, stream);
    launch_cvt(head_w, headw_h, NCLS * DMODEL, stream);
    launch_cvt(patch_w,patchw_h,DMODEL * 768, stream);
    launch_cvt(x,      xhp,     BATCH * 3 * 224 * 224, stream);

    k_init<<<1, 1, 0, stream>>>(sc);
    k_cls_init<<<BATCH, 128, 0, stream>>>(cls_tok, pos, xt0);
    {
        int waves = BATCH * 13 * 6;
        k_patch_embed<<<(waves * 32 + 255) / 256, 256, 0, stream>>>(xhp, patchw_h, patch_b, pos, xt0);
    }

    size_t rpk_total  = (size_t)BATCH * TCAP * 1152;
    int    rpk_blocks = (int)((rpk_total + 255) / 256);

    for (int l = 0; l < LNUM; ++l) {
        float* src = (l & 1) ? xt1 : xt0;
        float* dst = (l & 1) ? xt0 : xt1;

        // ---- pruning decision (pre-gather): ln1 + qkv + importance -----
        k_ln<<<Mrows, 128, 0, stream>>>(src, ln1_w + l * DMODEL, ln1_b + l * DMODEL,
                                        xnh, DMODEL, DMODEL);
        launch_gemm<0,1,1>(xnh, qkvw_h + (size_t)l * 1152 * DMODEL, qkv_b + l * 1152,
                           nullptr, bigh, Mrows, 1152, DMODEL, stream);
        k_repack<<<rpk_blocks, 256, 0, stream>>>(bigh, qfp, kfp, vtp);
        k_importance<<<BATCH, 256, 0, stream>>>(qfp, kfp, vtp, imp, &sc->n_tok);
        k_prune<<<1, 256, 0, stream>>>(imp, keep, sc);
        k_gather<<<dim3(BATCH, TCAP), 128, 0, stream>>>(src, dst, keep, sc);
        k_commit<<<1, 1, 0, stream>>>(sc);

        // ---- main path (post-gather) ----------------------------------
        k_ln<<<Mrows, 128, 0, stream>>>(dst, ln1_w + l * DMODEL, ln1_b + l * DMODEL,
                                        xnh, DMODEL, DMODEL);
        launch_gemm<0,1,1>(xnh, qkvw_h + (size_t)l * 1152 * DMODEL, qkv_b + l * 1152,
                           nullptr, bigh, Mrows, 1152, DMODEL, stream);
        k_repack<<<rpk_blocks, 256, 0, stream>>>(bigh, qfp, kfp, vtp);
        k_attn<<<dim3(BATCH * NHEAD, 13), 32, 0, stream>>>(qfp, kfp, vtp, oh, &sc->n_tok);
        launch_gemm<0,0,1>(oh, projw_h + (size_t)l * DMODEL * DMODEL, proj_b + l * DMODEL,
                           dst, dst, Mrows, DMODEL, DMODEL, stream);

        k_ln<<<Mrows, 128, 0, stream>>>(dst, ln2_w + l * DMODEL, ln2_b + l * DMODEL,
                                        xnh, DMODEL, DMODEL);
        launch_gemm<1,1,1>(xnh, fc1w_h + (size_t)l * 1536 * DMODEL, fc1_b + l * 1536,
                           nullptr, bigh, Mrows, 1536, DMODEL, stream);
        launch_gemm<0,0,1>(bigh, fc2w_h + (size_t)l * DMODEL * 1536, fc2_b + l * DMODEL,
                           dst, dst, Mrows, DMODEL, 1536, stream);
    }

    // after 12 layers, tokens are in xt0 (even number of ping-pongs)
    k_ln<<<BATCH, 128, 0, stream>>>(xt0, norm_w, norm_b, clsn,
                                    TCAP * DMODEL, DMODEL);
    launch_gemm<0,0,0>(clsn, headw_h, head_b, nullptr, out, BATCH, NCLS, DMODEL, stream);
}
